// TimeGrad_9423158247523
// MI455X (gfx1250) — matrix operand order repaired and checked by reading, not compiled, not run on hardware
//
#include <hip/hip_runtime.h>
#include <hip/hip_bf16.h>
#include <math.h>

// ---------------- types ----------------
typedef __bf16 bf16_t;
typedef bf16_t v16bf __attribute__((ext_vector_type(16)));
typedef bf16_t v8bf  __attribute__((ext_vector_type(8)));
typedef float  v8f   __attribute__((ext_vector_type(8)));

#define BATCH     16384
#define TT        64
#define IN_DIM    32
#define HDIM      64
#define DDIM      32
#define UHDIM     128
#define TEMB_DIM  64
#define NUM_INF   10
#define PRED      4

#define WAVES         4
#define ROWS_PER_WAVE 16
#define ROWS_PER_WG   (WAVES * ROWS_PER_WAVE)   // 64
#define NWG           (BATCH / ROWS_PER_WG)     // 256

struct Sched {
  float ts[NUM_INF], sab[NUM_INF], s1m[NUM_INF];
  float c0[NUM_INF], cx[NUM_INF], sig[NUM_INF];
};

// ---------------- helpers ----------------
__device__ __forceinline__ bf16_t f2bf(float f) {
#if defined(__gfx1250__)
  return (bf16_t)f;                     // native v_cvt_pk_bf16_f32 on CDNA5
#else
  unsigned u = __builtin_bit_cast(unsigned, f);
  unsigned r = (u + 0x7FFFu + ((u >> 16) & 1u)) >> 16;   // RNE
  return __builtin_bit_cast(bf16_t, (unsigned short)r);
#endif
}
__device__ __forceinline__ float sigm(float x) { return 1.f / (1.f + __expf(-x)); }
__device__ __forceinline__ float tanh_fast(float x) { return 2.f / (1.f + __expf(-2.f * x)) - 1.f; }

// scheduling fence: DS and WMMA must not cross; ALU/VALU/SALU/transcendental may.
__device__ __forceinline__ void sched_fence() {
#if __has_builtin(__builtin_amdgcn_sched_barrier)
  __builtin_amdgcn_sched_barrier(0x407);
#endif
}

__device__ __forceinline__ unsigned hashu(unsigned x) {
  x ^= x >> 16; x *= 0x7feb352du; x ^= x >> 15; x *= 0x846ca68bu; x ^= x >> 16; return x;
}
// deterministic counter-based standard normal (Box-Muller)
__device__ __forceinline__ float gauss(unsigned idx) {
  unsigned a = hashu(idx * 2u + 0x9E3779B9u);
  unsigned b = hashu(idx * 2u + 1u + 0x85EBCA6Bu);
  float u1 = (float)(a >> 8) * (1.f / 16777216.f) + 1.0e-7f;
  float u2 = (float)(b >> 8) * (1.f / 16777216.f);
  return sqrtf(-2.f * __logf(u1)) * __cosf(6.28318530718f * u2);
}

__device__ __forceinline__ v8f splat8(float v) {
  v8f r;
#pragma unroll
  for (int i = 0; i < 8; i++) r[i] = v;
  return r;
}

__device__ __forceinline__ v8f wmma_bf(v16bf a, v16bf b, v8f c) {
  return __builtin_amdgcn_wmma_f32_16x16x32_bf16(false, a, false, b, (short)0, c, false, false);
}

// B fragment (K=32 x N=16) stored pre-swizzled in LDS: lane -> 16 contiguous bf16
__device__ __forceinline__ v16bf loadBfrag(const bf16_t* base, int frag, int lane) {
  return *(const v16bf*)(base + (frag << 9) + (lane << 4));
}

// A fragment (M=16 x K=32) from row-major bf16 LDS stage; rowlen = K dim of stage
__device__ __forceinline__ v16bf loadAfrag(const bf16_t* st, int rowlen, int kh, int lane) {
  int m = lane & 15;
  int kb = (lane >> 4) << 3;
  const bf16_t* p = st + m * rowlen + kh * 32 + kb;
  v8bf lo = *(const v8bf*)p;
  v8bf hi = *(const v8bf*)(p + 16);
  v16bf r;
#pragma unroll
  for (int i = 0; i < 8; i++) { r[i] = lo[i]; r[i + 8] = hi[i]; }
  return r;
}

// A fragment directly from global f32 x[b, t, :] rows (K = 32)
__device__ __forceinline__ v16bf loadXfrag(const float* __restrict__ X, int rowBase, int t, int lane) {
  int m = lane & 15;
  int kb = (lane >> 4) << 3;
  const float* p = X + ((size_t)(rowBase + m) * TT + t) * IN_DIM;
  v16bf r;
#pragma unroll
  for (int i = 0; i < 8; i++) {
    r[i]     = f2bf(p[kb + i]);
    r[i + 8] = f2bf(p[16 + kb + i]);
  }
  return r;
}

// convert a weight matrix (f32 global) into swizzled bf16 B-fragments in LDS.
// B[k,n] = trans ? W[n*Kdim+k] : W[k*Ndim+n]; frag index = fn*KF + fk.
__device__ void fillB(bf16_t* dst, const float* __restrict__ W, int Kdim, int Ndim, bool trans) {
  int KF = Kdim >> 5;
  int total = Kdim * Ndim;
  for (int h = threadIdx.x; h < total; h += blockDim.x) {
    int frag = h >> 9;
    int rem = h & 511;
    int ln = rem >> 4, e = rem & 15;
    int fk = frag % KF, fn = frag / KF;
    int k = fk * 32 + ((e >> 3) << 4) + ((ln >> 4) << 3) + (e & 7);
    int n = fn * 16 + (ln & 15);
    float w = trans ? W[n * Kdim + k] : W[k * Ndim + n];
    dst[h] = f2bf(w);
  }
}

// one fused LSTM step for both layers on this wave's 16 rows.
// h state lives in h0st/h1st (row-major bf16), c state in registers (C-layout).
// biases live in per-lane registers and are folded into the activation chain
// (leading WMMAs start from the shared zero accumulator z8).
__device__ __forceinline__ void lstm_step(
    v16bf xA, v8f z8,
    bf16_t* h0st, bf16_t* h1st,
    const bf16_t* Wih0f, const bf16_t* Whh0f,
    const bf16_t* Wih1f, const bf16_t* Whh1f,
    const float (&b0r)[16], const float (&b1r)[16],
    v8f (&c0s)[4], v8f (&c1s)[4], int lane) {
  int col0 = lane & 15;
  int rowoff = (lane >> 4) << 3;

  v16bf h0A0 = loadAfrag(h0st, HDIM, 0, lane);
  v16bf h0A1 = loadAfrag(h0st, HDIM, 1, lane);
  v16bf h1A0 = loadAfrag(h1st, HDIM, 0, lane);
  v16bf h1A1 = loadAfrag(h1st, HDIM, 1, lane);

  // ----- layer 0 -----
#pragma unroll
  for (int th = 0; th < 4; ++th) {
    v16bf w0[4], w1[4], w2[4];
#pragma unroll
    for (int gate = 0; gate < 4; ++gate) {
      int tile = gate * 4 + th;
      w0[gate] = loadBfrag(Wih0f, tile,         lane);
      w1[gate] = loadBfrag(Whh0f, tile * 2 + 0, lane);
      w2[gate] = loadBfrag(Whh0f, tile * 2 + 1, lane);
    }
    sched_fence();
    v8f g[4];
#pragma unroll
    for (int gate = 0; gate < 4; ++gate) g[gate] = wmma_bf(xA, w0[gate], z8);
#pragma unroll
    for (int gate = 0; gate < 4; ++gate) g[gate] = wmma_bf(h0A0, w1[gate], g[gate]);
#pragma unroll
    for (int gate = 0; gate < 4; ++gate) g[gate] = wmma_bf(h0A1, w2[gate], g[gate]);
#pragma unroll
    for (int r = 0; r < 8; ++r) {
      float iv = sigm(g[0][r] + b0r[th]);
      float fv = sigm(g[1][r] + b0r[4 + th]);
      float gv = tanh_fast(g[2][r] + b0r[8 + th]);
      float ov = sigm(g[3][r] + b0r[12 + th]);
      float c = fv * c0s[th][r] + iv * gv;
      c0s[th][r] = c;
      float h = ov * tanh_fast(c);
      h0st[(r + rowoff) * HDIM + th * 16 + col0] = f2bf(h);
    }
  }
  v16bf n0A0 = loadAfrag(h0st, HDIM, 0, lane);
  v16bf n0A1 = loadAfrag(h0st, HDIM, 1, lane);

  // ----- layer 1 -----
#pragma unroll
  for (int th = 0; th < 4; ++th) {
    v16bf w0[4], w1[4], w2[4], w3[4];
#pragma unroll
    for (int gate = 0; gate < 4; ++gate) {
      int tile = gate * 4 + th;
      w0[gate] = loadBfrag(Wih1f, tile * 2 + 0, lane);
      w1[gate] = loadBfrag(Wih1f, tile * 2 + 1, lane);
      w2[gate] = loadBfrag(Whh1f, tile * 2 + 0, lane);
      w3[gate] = loadBfrag(Whh1f, tile * 2 + 1, lane);
    }
    sched_fence();
    v8f g[4];
#pragma unroll
    for (int gate = 0; gate < 4; ++gate) g[gate] = wmma_bf(n0A0, w0[gate], z8);
#pragma unroll
    for (int gate = 0; gate < 4; ++gate) g[gate] = wmma_bf(n0A1, w1[gate], g[gate]);
#pragma unroll
    for (int gate = 0; gate < 4; ++gate) g[gate] = wmma_bf(h1A0, w2[gate], g[gate]);
#pragma unroll
    for (int gate = 0; gate < 4; ++gate) g[gate] = wmma_bf(h1A1, w3[gate], g[gate]);
#pragma unroll
    for (int r = 0; r < 8; ++r) {
      float iv = sigm(g[0][r] + b1r[th]);
      float fv = sigm(g[1][r] + b1r[4 + th]);
      float gv = tanh_fast(g[2][r] + b1r[8 + th]);
      float ov = sigm(g[3][r] + b1r[12 + th]);
      float c = fv * c1s[th][r] + iv * gv;
      c1s[th][r] = c;
      float h = ov * tanh_fast(c);
      h1st[(r + rowoff) * HDIM + th * 16 + col0] = f2bf(h);
    }
  }
}

// ---------------- fused persistent kernel ----------------
__global__ void __launch_bounds__(128)
timegrad_fused(const float* __restrict__ X,
               const float* __restrict__ Wih0, const float* __restrict__ Whh0, const float* __restrict__ b0,
               const float* __restrict__ Wih1, const float* __restrict__ Whh1, const float* __restrict__ b1,
               const float* __restrict__ Wx, const float* __restrict__ Wc, const float* __restrict__ Wt,
               const float* __restrict__ bu1, const float* __restrict__ W2, const float* __restrict__ bu2,
               const float* __restrict__ Wo, const float* __restrict__ bo,
               float* __restrict__ Out, Sched sc) {
  extern __shared__ __align__(16) unsigned char smemraw[];
  // weight fragments (bf16 halves)
  bf16_t* Wih0f = (bf16_t*)smemraw;       // 32*256  =  8192
  bf16_t* Whh0f = Wih0f + 8192;           // 64*256  = 16384
  bf16_t* Wih1f = Whh0f + 16384;          // 64*256  = 16384
  bf16_t* Whh1f = Wih1f + 16384;          // 64*256  = 16384
  bf16_t* Wxf   = Whh1f + 16384;          // 32*128  =  4096
  bf16_t* Wcf   = Wxf   + 4096;           // 64*128  =  8192
  bf16_t* W2f   = Wcf   + 8192;           // 128*128 = 16384
  bf16_t* Wof   = W2f   + 16384;          // 128*32  =  4096
  bf16_t* stages = Wof  + 4096;           // 4 waves * 6656 halves
  float*  fsec  = (float*)(stages + WAVES * 6656);
  float* b0s   = fsec;            // 256
  float* b1s   = b0s + 256;       // 256
  float* bu2s  = b1s + 256;       // 128
  float* bos   = bu2s + 128;      // 32
  float* tvecs = bos + 32;        // NUM_INF * 128

  // ---- one-time fill: weights -> swizzled bf16 fragments ----
  fillB(Wih0f, Wih0, 32, 256, true);
  fillB(Whh0f, Whh0, 64, 256, true);
  fillB(Wih1f, Wih1, 64, 256, true);
  fillB(Whh1f, Whh1, 64, 256, true);
  fillB(Wxf, Wx, 32, 128, false);
  fillB(Wcf, Wc, 64, 128, false);
  fillB(W2f, W2, 128, 128, false);
  fillB(Wof, Wo, 128, 32, false);
  for (int i = threadIdx.x; i < 256; i += blockDim.x) { b0s[i] = b0[i]; b1s[i] = b1[i]; }
  for (int i = threadIdx.x; i < 128; i += blockDim.x) bu2s[i] = bu2[i];
  for (int i = threadIdx.x; i < 32; i += blockDim.x) bos[i] = bo[i];
  // timestep embedding projected through Wt + bu1, once per inference step
  for (int s = 0; s < NUM_INF; ++s) {
    float t = sc.ts[s];
    for (int n = threadIdx.x; n < UHDIM; n += blockDim.x) {
      float acc = bu1[n];
      for (int k = 0; k < TEMB_DIM; ++k) {
        float fr = __expf(-__logf(10000.f) * (float)(k & 31) * (1.f / 32.f));
        float ang = t * fr;
        float e = (k < 32) ? __sinf(ang) : __cosf(ang);
        acc += e * Wt[k * UHDIM + n];
      }
      tvecs[s * UHDIM + n] = acc;
    }
  }
  __syncthreads();

  int wave = threadIdx.x >> 5;
  int lane = threadIdx.x & 31;
  int col0 = lane & 15;
  int rowoff = (lane >> 4) << 3;
  int rowBase = blockIdx.x * ROWS_PER_WG + wave * ROWS_PER_WAVE;

  bf16_t* h0st = stages + wave * 6656;  // 16x64
  bf16_t* h1st = h0st + 1024;           // 16x64  (== cond for diffusion)
  bf16_t* stA  = h1st + 1024;           // 16x128 (h1)
  bf16_t* stB  = stA + 2048;            // 16x128 (h2)
  bf16_t* xst  = stB + 2048;            // 16x32  (diffusion x / AR input)

  bf16_t zbf = f2bf(0.f);
  for (int i = lane; i < 1024; i += 32) { h0st[i] = zbf; h1st[i] = zbf; }

  // loop-invariant per-lane biases -> registers (no LDS traffic in hot loops)
  float b0r[16], b1r[16];
#pragma unroll
  for (int tile = 0; tile < 16; ++tile) {
    b0r[tile] = b0s[tile * 16 + col0];
    b1r[tile] = b1s[tile * 16 + col0];
  }
  float bu2r[8], bor[2];
#pragma unroll
  for (int n = 0; n < 8; ++n) bu2r[n] = bu2s[n * 16 + col0];
#pragma unroll
  for (int n = 0; n < 2; ++n) bor[n] = bos[n * 16 + col0];

  v8f z8 = splat8(0.f);   // shared zero accumulator for all leading WMMAs
  v8f c0s[4], c1s[4];
#pragma unroll
  for (int th = 0; th < 4; ++th) { c0s[th] = z8; c1s[th] = z8; }

  // ---- main LSTM over T (prefetch x for t+1 behind the WMMA work) ----
  for (int t = 0; t < TT; ++t) {
    if (t + 1 < TT) {
      const float* pn = X + ((size_t)(rowBase + (lane & 15)) * TT + (t + 1)) * IN_DIM
                          + ((lane >> 4) << 4);
      __builtin_prefetch(pn, 0, 0);     // global_prefetch_b8
    }
    v16bf xA = loadXfrag(X, rowBase, t, lane);
    lstm_step(xA, z8, h0st, h1st, Wih0f, Whh0f, Wih1f, Whh1f, b0r, b1r, c0s, c1s, lane);
  }

  // ---- autoregressive prediction: DDPM sampling + 1-step LSTM ----
  for (int p = 0; p < PRED; ++p) {
    v8f xacc[2];
#pragma unroll
    for (int n = 0; n < 2; ++n)
#pragma unroll
      for (int r = 0; r < 8; ++r) {
        int b = rowBase + r + rowoff;
        int d = n * 16 + col0;
        unsigned idx = ((unsigned)(p * (NUM_INF + 1)) * BATCH + (unsigned)b) * DDIM + (unsigned)d;
        xacc[n][r] = gauss(idx);
      }

    for (int s = 0; s < NUM_INF; ++s) {
      // stage diffusion state x (C-layout regs -> row-major bf16)
#pragma unroll
      for (int n = 0; n < 2; ++n)
#pragma unroll
        for (int r = 0; r < 8; ++r)
          xst[(r + rowoff) * DDIM + n * 16 + col0] = f2bf(xacc[n][r]);

      v16bf xdA = loadAfrag(xst, DDIM, 0, lane);
      v16bf cA0 = loadAfrag(h1st, HDIM, 0, lane);
      v16bf cA1 = loadAfrag(h1st, HDIM, 1, lane);
      float tvr[8];
#pragma unroll
      for (int n = 0; n < 8; ++n) tvr[n] = tvecs[s * UHDIM + n * 16 + col0];

      // h1 = relu(x@Wx + cond@Wc + tvec); WMMAs grouped by A operand
      v8f a[8];
      {
        v16bf w[8];
#pragma unroll
        for (int n = 0; n < 8; ++n) w[n] = loadBfrag(Wxf, n, lane);
        sched_fence();
#pragma unroll
        for (int n = 0; n < 8; ++n) a[n] = wmma_bf(xdA, w[n], z8);
      }
      {
        v16bf w[8];
#pragma unroll
        for (int n = 0; n < 8; ++n) w[n] = loadBfrag(Wcf, n * 2 + 0, lane);
        sched_fence();
#pragma unroll
        for (int n = 0; n < 8; ++n) a[n] = wmma_bf(cA0, w[n], a[n]);
      }
      {
        v16bf w[8];
#pragma unroll
        for (int n = 0; n < 8; ++n) w[n] = loadBfrag(Wcf, n * 2 + 1, lane);
        sched_fence();
#pragma unroll
        for (int n = 0; n < 8; ++n) a[n] = wmma_bf(cA1, w[n], a[n]);
      }
#pragma unroll
      for (int n = 0; n < 8; ++n)
#pragma unroll
        for (int r = 0; r < 8; ++r) {
          float v = a[n][r] + tvr[n]; v = v > 0.f ? v : 0.f;
          stA[(r + rowoff) * UHDIM + n * 16 + col0] = f2bf(v);
        }

      v16bf hA[4];
#pragma unroll
      for (int kf = 0; kf < 4; ++kf) hA[kf] = loadAfrag(stA, UHDIM, kf, lane);

      // h2 = relu(h1@W2 + bu2)
#pragma unroll
      for (int kf = 0; kf < 4; ++kf) {
        v16bf w[8];
#pragma unroll
        for (int n = 0; n < 8; ++n) w[n] = loadBfrag(W2f, n * 4 + kf, lane);
        sched_fence();
        if (kf == 0) {
#pragma unroll
          for (int n = 0; n < 8; ++n) a[n] = wmma_bf(hA[0], w[n], z8);
        } else {
#pragma unroll
          for (int n = 0; n < 8; ++n) a[n] = wmma_bf(hA[kf], w[n], a[n]);
        }
      }
#pragma unroll
      for (int n = 0; n < 8; ++n)
#pragma unroll
        for (int r = 0; r < 8; ++r) {
          float v = a[n][r] + bu2r[n]; v = v > 0.f ? v : 0.f;
          stB[(r + rowoff) * UHDIM + n * 16 + col0] = f2bf(v);
        }

      v16bf gA[4];
#pragma unroll
      for (int kf = 0; kf < 4; ++kf) gA[kf] = loadAfrag(stB, UHDIM, kf, lane);

      // eps = h2@Wo + bo ; DDPM update
      float sa = sc.sab[s], s1 = sc.s1m[s];
      float cc0 = sc.c0[s], ccx = sc.cx[s], sg = sc.sig[s];
      float inv_sa = 1.f / sa;
      v8f e[2];
      {
        v16bf w[8];
#pragma unroll
        for (int kf = 0; kf < 4; ++kf)
#pragma unroll
          for (int n = 0; n < 2; ++n) w[kf * 2 + n] = loadBfrag(Wof, n * 4 + kf, lane);
        sched_fence();
#pragma unroll
        for (int n = 0; n < 2; ++n) e[n] = wmma_bf(gA[0], w[n], z8);
#pragma unroll
        for (int kf = 1; kf < 4; ++kf)
#pragma unroll
          for (int n = 0; n < 2; ++n) e[n] = wmma_bf(gA[kf], w[kf * 2 + n], e[n]);
      }
#pragma unroll
      for (int n = 0; n < 2; ++n)
#pragma unroll
        for (int r = 0; r < 8; ++r) {
          float xv = xacc[n][r];
          float px0 = (xv - s1 * (e[n][r] + bor[n])) * inv_sa;
          int b = rowBase + r + rowoff;
          int d = n * 16 + col0;
          unsigned idx = ((unsigned)(p * (NUM_INF + 1) + 1 + s) * BATCH + (unsigned)b) * DDIM + (unsigned)d;
          float z = gauss(idx);
          xacc[n][r] = cc0 * px0 + ccx * xv + sg * z;
        }
    }

    // write sample to output [B, PRED, D]
#pragma unroll
    for (int n = 0; n < 2; ++n)
#pragma unroll
      for (int r = 0; r < 8; ++r) {
        int b = rowBase + r + rowoff;
        int d = n * 16 + col0;
        Out[((size_t)b * PRED + p) * DDIM + d] = xacc[n][r];
      }

    // feed sample back through one LSTM step (cond for next prediction)
    if (p < PRED - 1) {
#pragma unroll
      for (int n = 0; n < 2; ++n)
#pragma unroll
        for (int r = 0; r < 8; ++r)
          xst[(r + rowoff) * DDIM + n * 16 + col0] = f2bf(xacc[n][r]);
      v16bf xA = loadAfrag(xst, DDIM, 0, lane);
      lstm_step(xA, z8, h0st, h1st, Wih0f, Whh0f, Wih1f, Whh1f, b0r, b1r, c0s, c1s, lane);
    }
  }
}

// ---------------- host launcher ----------------
extern "C" void kernel_launch(void* const* d_in, const int* in_sizes, int n_in,
                              void* d_out, int out_size, void* d_ws, size_t ws_size,
                              hipStream_t stream) {
  (void)in_sizes; (void)n_in; (void)out_size; (void)d_ws; (void)ws_size;

  // DDPM schedule (matches reference _make_sched), computed in double on host
  Sched sc;
  {
    double ab[1000];
    double c = 1.0;
    for (int i = 0; i < 1000; ++i) {
      double beta = 1e-4 + (0.02 - 1e-4) * (double)i / 999.0;
      c *= (1.0 - beta);
      ab[i] = c;
    }
    const int ratio = 1000 / NUM_INF;
    for (int s = 0; s < NUM_INF; ++s) {
      int t = (NUM_INF - 1 - s) * ratio;   // descending timesteps
      int prev = t - ratio;
      double ab_t = ab[t];
      double ab_p = (prev >= 0) ? ab[prev] : 1.0;
      double a_cur = ab_t / ab_p, b_cur = 1.0 - a_cur;
      double c0v = sqrt(ab_p) * b_cur / (1.0 - ab_t);
      double cxv = sqrt(a_cur) * (1.0 - ab_p) / (1.0 - ab_t);
      double var = (1.0 - ab_p) / (1.0 - ab_t) * b_cur;
      if (var < 1e-20) var = 1e-20;
      double sg = (prev >= 0) ? sqrt(var) : 0.0;
      sc.ts[s]  = (float)t;
      sc.sab[s] = (float)sqrt(ab_t);
      sc.s1m[s] = (float)sqrt(1.0 - ab_t);
      sc.c0[s]  = (float)c0v;
      sc.cx[s]  = (float)cxv;
      sc.sig[s] = (float)sg;
    }
  }

  // dynamic LDS: 90112 weight halves + 4*6656 stage halves + 1952 floats
  size_t smem = (size_t)(90112 + WAVES * 6656) * sizeof(unsigned short) + 1952 * sizeof(float);
  (void)hipFuncSetAttribute((const void*)timegrad_fused,
                            hipFuncAttributeMaxDynamicSharedMemorySize, (int)smem);

  timegrad_fused<<<dim3(NWG), dim3(WAVES * 32), smem, stream>>>(
      (const float*)d_in[0],
      (const float*)d_in[1], (const float*)d_in[2], (const float*)d_in[3],
      (const float*)d_in[4], (const float*)d_in[5], (const float*)d_in[6],
      (const float*)d_in[7], (const float*)d_in[8], (const float*)d_in[9],
      (const float*)d_in[10], (const float*)d_in[11], (const float*)d_in[12],
      (const float*)d_in[13], (const float*)d_in[14],
      (float*)d_out, sc);
}
